// AdaClusteringAttention_17197049053474
// MI455X (gfx1250) — matrix operand order repaired and checked by reading, not compiled, not run on hardware
//
#include <hip/hip_runtime.h>

// ---------------- problem constants (match reference) ----------------
constexpr int cB0 = 8;              // raw batch
constexpr int cN  = 4096;           // seq len
constexpr int cD  = 128;            // head dim
constexpr int cB  = 64;             // B0*H flattened
constexpr int cC  = 513;            // clusters + 1
constexpr int cCp = 544;            // padded C for transposed V (multiple of 32, >= 544)

// ---------------- WMMA types ----------------
typedef __attribute__((ext_vector_type(16))) __bf16 v16bf;
typedef __attribute__((ext_vector_type(8)))  float  v8f;

union Frag16 {
    v16bf v;
    uint4 q[2];
    unsigned int u[8];
};

static __device__ inline unsigned short f2bf(float f) {
    unsigned int u = __float_as_uint(f);
    unsigned int r = (u + 0x7fffu + ((u >> 16) & 1u)) >> 16;   // RNE
    return (unsigned short)r;
}
static __device__ inline v8f wmma_bf16(v16bf a, v16bf b, v8f c) {
    return __builtin_amdgcn_wmma_f32_16x16x32_bf16(false, a, false, b, (short)0, c, false, false);
}

// ---------------- phase 1: histogram / buckets ----------------
__global__ void zero_i32_kernel(int* p, int n) {
    int i = blockIdx.x * blockDim.x + threadIdx.x;
    if (i < n) p[i] = 0;
}

__global__ void count_kernel(const int* __restrict__ clusters, int* __restrict__ counts) {
    int i = blockIdx.x * blockDim.x + threadIdx.x;       // over B0*N
    if (i >= cB0 * cN) return;
    int b0 = i >> 12;                                     // / 4096
    int cl = clusters[i];
    atomicAdd(&counts[b0 * cC + cl], 1);
}

__global__ void scan_kernel(const int* __restrict__ counts, int* __restrict__ offs,
                            int* __restrict__ cursor, float* __restrict__ countsF) {
    int b0 = threadIdx.x;
    if (b0 >= cB0) return;
    int run = 0;
    for (int c = 0; c < cC; ++c) {
        int cc = counts[b0 * cC + c];
        offs[b0 * cC + c]   = run;
        cursor[b0 * cC + c] = run;
        countsF[b0 * cC + c] = (float)cc;
        run += cc;
    }
}

__global__ void scatter_kernel(const int* __restrict__ clusters, int* __restrict__ cursor,
                               int* __restrict__ bucket) {
    int i = blockIdx.x * blockDim.x + threadIdx.x;       // over B0*N
    if (i >= cB0 * cN) return;
    int b0 = i >> 12;
    int n  = i & (cN - 1);
    int cl = clusters[i];
    int slot = atomicAdd(&cursor[b0 * cC + cl], 1);      // set deterministic; order fixed by sort
    bucket[b0 * cN + slot] = n;
}

// zero the padded tail columns of transposed V (e in [cC, cCp))
__global__ void padzero_kernel(unsigned short* __restrict__ vcT) {
    int i = blockIdx.x * blockDim.x + threadIdx.x;       // over B*D*(cCp-cC)
    constexpr int PAD = cCp - cC;                         // 31
    if (i >= cB * cD * PAD) return;
    int cpad = i % PAD;
    int bd   = i / PAD;                                   // b*cD + d
    vcT[(size_t)bd * cCp + cC + cpad] = 0;
}

// ---------------- phase 2: cluster means -> bf16 centers ----------------
__global__ __launch_bounds__(128)
void center_kernel(const float* __restrict__ q, const float* __restrict__ k,
                   const float* __restrict__ v, const int* __restrict__ counts,
                   const int* __restrict__ offs, const int* __restrict__ bucket,
                   unsigned short* __restrict__ qcH, unsigned short* __restrict__ kcH,
                   unsigned short* __restrict__ vcT) {
    int c = blockIdx.x;          // cluster
    int b = blockIdx.y;          // flattened batch
    int b0 = b & (cB0 - 1);
    int cnt   = counts[b0 * cC + c];
    int start = offs[b0 * cC + c];

    __shared__ int toks[32];
    __shared__ int scnt;
    if (threadIdx.x == 0) {
        int m = cnt > 32 ? 32 : cnt;
        for (int i = 0; i < m; ++i) toks[i] = bucket[b0 * cN + start + i];
        // insertion sort -> canonical (ascending token) summation order
        for (int i = 1; i < m; ++i) {
            int key = toks[i]; int j = i - 1;
            while (j >= 0 && toks[j] > key) { toks[j + 1] = toks[j]; --j; }
            toks[j + 1] = key;
        }
        scnt = m;
    }
    __syncthreads();

    int d = threadIdx.x;         // 0..127
    float sq = 0.f, sk = 0.f, sv = 0.f;
    int m = scnt;
    for (int i = 0; i < m; ++i) {
        size_t off = ((size_t)b * cN + toks[i]) * cD + d;
        sq += q[off]; sk += k[off]; sv += v[off];
    }
    float w = cnt > 0 ? 1.0f / (float)cnt : 0.0f;
    size_t o = ((size_t)b * cC + c) * cD + d;
    qcH[o] = f2bf(sq * w);
    kcH[o] = f2bf(sk * w);
    vcT[((size_t)b * cD + d) * cCp + c] = f2bf(sv * w);
}

// ---------------- phase 3: clustered attention (WMMA, one wave / 16 rows) ----------------
__global__ __launch_bounds__(32)
void attn_kernel(const unsigned short* __restrict__ qcH, const unsigned short* __restrict__ kcH,
                 const unsigned short* __restrict__ vcT, const float* __restrict__ countsF,
                 float* __restrict__ vout, float* __restrict__ a0out) {
    const int tile = blockIdx.x;           // 16-row tile of clusters (33 tiles)
    const int b    = blockIdx.y;           // batch
    const int lane = threadIdx.x;
    const int n16  = lane & 15;
    const int half = lane >> 4;

    const unsigned short* qb = qcH + (size_t)b * cC * cD;
    const unsigned short* kb = kcH + (size_t)b * cC * cD;
    const unsigned short* vb = vcT + (size_t)b * cD * cCp;
    const float* cw = countsF + (size_t)(b & (cB0 - 1)) * cC;

    const int rowbase = tile * 16;
    int qrow = rowbase + n16; if (qrow >= cC) qrow = cC - 1;   // clamp for load safety

    // --- Q A-fragments: contiguous bf16 (A layout: M=lane%16, K=half*16+2j) ---
    Frag16 aq[4];
    for (int c = 0; c < 4; ++c) {
        const unsigned short* qp = qb + (size_t)qrow * cD + c * 32;
        aq[c].q[0] = *(const uint4*)(qp + half * 16);          // K = half*16 + 0..7
        aq[c].q[1] = *(const uint4*)(qp + half * 16 + 8);      // K = half*16 + 8..15
    }

    v8f acc[8];
    for (int t = 0; t < 8; ++t)
        for (int r = 0; r < 8; ++r) acc[t][r] = 0.0f;
    float m_[8], l_[8], p0_[8];
    for (int r = 0; r < 8; ++r) { m_[r] = -1e30f; l_[r] = 0.0f; p0_[r] = 0.0f; }

    __shared__ unsigned short ldsP[16 * 32];   // P tile staging (bf16, row-major 16x32)

    for (int et = 0; et < 17; ++et) {          // 32-wide key tiles over C=513
        const int e0 = et * 32;
        const int ecol0 = e0 + n16;
        const int ecol1 = e0 + 16 + n16;
        const int le0 = ecol0 < cC ? ecol0 : cC - 1;
        const int le1 = ecol1 < cC ? ecol1 : cC - 1;
        const bool v0 = ecol0 < cC, v1 = ecol1 < cC;

        // predicated (branch-free) count-weight loads from clamped index
        float w0 = cw[le0]; if (!v0) w0 = 0.0f;
        float w1 = cw[le1]; if (!v1) w1 = 0.0f;

        // --- S = Qc * Kc^T : two 16x16 tiles, double-buffered K fragments ---
        v8f s0, s1;
        for (int r = 0; r < 8; ++r) { s0[r] = 0.0f; s1[r] = 0.0f; }
        const unsigned short* kp0 = kb + (size_t)le0 * cD + half * 16;  // B layout: K=half*16+2j
        const unsigned short* kp1 = kb + (size_t)le1 * cD + half * 16;
        Frag16 bk0[2], bk1[2];
        bk0[0].q[0] = *(const uint4*)(kp0);
        bk0[0].q[1] = *(const uint4*)(kp0 + 8);
        bk1[0].q[0] = *(const uint4*)(kp1);
        bk1[0].q[1] = *(const uint4*)(kp1 + 8);
#pragma unroll
        for (int c = 0; c < 4; ++c) {
            const int cur = c & 1, nxt = cur ^ 1;
            if (c < 3) {                                  // prefetch next chunk
                bk0[nxt].q[0] = *(const uint4*)(kp0 + (c + 1) * 32);
                bk0[nxt].q[1] = *(const uint4*)(kp0 + (c + 1) * 32 + 8);
                bk1[nxt].q[0] = *(const uint4*)(kp1 + (c + 1) * 32);
                bk1[nxt].q[1] = *(const uint4*)(kp1 + (c + 1) * 32 + 8);
            }
            s0 = wmma_bf16(aq[c].v, bk0[cur].v, s0);
            s1 = wmma_bf16(aq[c].v, bk1[cur].v, s1);
        }

        // --- online count-weighted softmax (row = half*8 + r) ---
        float scl[8];
        for (int r = 0; r < 8; ++r) {
            float sv0 = v0 ? s0[r] : -1e30f;
            float sv1 = v1 ? s1[r] : -1e30f;
            float smax = fmaxf(sv0, sv1);
            smax = fmaxf(smax, __shfl_xor(smax, 1, 32));
            smax = fmaxf(smax, __shfl_xor(smax, 2, 32));
            smax = fmaxf(smax, __shfl_xor(smax, 4, 32));
            smax = fmaxf(smax, __shfl_xor(smax, 8, 32));
            float mnew = fmaxf(m_[r], smax);
            float sc   = __expf(m_[r] - mnew);
            float pv0  = w0 * __expf(sv0 - mnew);
            float pv1  = w1 * __expf(sv1 - mnew);
            float rs = pv0 + pv1;
            rs += __shfl_xor(rs, 1, 32);
            rs += __shfl_xor(rs, 2, 32);
            rs += __shfl_xor(rs, 4, 32);
            rs += __shfl_xor(rs, 8, 32);
            l_[r] = l_[r] * sc + rs;
            m_[r] = mnew;
            p0_[r] = (et == 0) ? pv0 : p0_[r] * sc;   // track column 0 (valid in lane n16==0)
            scl[r] = sc;
            int prow = half * 8 + r;
            ldsP[prow * 32 + n16]      = f2bf(pv0);
            ldsP[prow * 32 + 16 + n16] = f2bf(pv1);
        }
        __syncthreads();

        // --- P (16x32) as A-fragment via LDS transpose ---
        Frag16 pf;
        for (int j = 0; j < 8; ++j) {
            int K = half * 16 + (j << 1);
            pf.u[j] = *(const unsigned int*)&ldsP[n16 * 32 + K];
        }

        // --- acc = acc*scale + P * Vtile : 8 bf16 WMMAs, double-buffered V fragments ---
        const unsigned short* vpb = vb + e0 + half * 16;      // + dcol*cCp per tile
        Frag16 bv[2];
        {
            const unsigned short* vp = vpb + (size_t)n16 * cCp;       // t=0: dcol=n16
            bv[0].q[0] = *(const uint4*)(vp);
            bv[0].q[1] = *(const uint4*)(vp + 8);
        }
#pragma unroll
        for (int t = 0; t < 8; ++t) {
            const int cur = t & 1, nxt = cur ^ 1;
            if (t < 7) {                                  // prefetch next d-tile
                const unsigned short* vp = vpb + (size_t)((t + 1) * 16 + n16) * cCp;
                bv[nxt].q[0] = *(const uint4*)(vp);
                bv[nxt].q[1] = *(const uint4*)(vp + 8);
            }
            for (int r = 0; r < 8; ++r) acc[t][r] *= scl[r];
            acc[t] = wmma_bf16(pf.v, bv[cur].v, acc[t]);
        }
        __syncthreads();   // protect ldsP reuse across iterations
    }

    // --- epilogue: normalize, store cluster outputs and a[:, :, 0] ---
    for (int r = 0; r < 8; ++r) {
        int row = rowbase + half * 8 + r;
        if (row < cC) {
            float inv = 1.0f / l_[r];
            size_t o = ((size_t)b * cC + row) * cD;
            for (int t = 0; t < 8; ++t)
                vout[o + t * 16 + n16] = acc[t][r] * inv;
            if (n16 == 0)
                a0out[(size_t)b * cC + row] = p0_[r] * inv;
        }
    }
}

// ---------------- phase 4: broadcast cluster outputs to tokens ----------------
__global__ __launch_bounds__(256)
void gather_kernel(const float* __restrict__ vout, const int* __restrict__ clusters,
                   float* __restrict__ out) {
    int i = blockIdx.x * blockDim.x + threadIdx.x;   // over B*N*(D/4) = 8.39M
    if (i >= cB * cN * (cD / 4)) return;
    int d4 = i & 31;                                  // D/4 = 32
    int bn = i >> 5;
    int n  = bn & (cN - 1);
    int b  = bn >> 12;
    int g  = clusters[(b & (cB0 - 1)) * cN + n];
    const float4* src = (const float4*)(vout + ((size_t)b * cC + g) * cD) + d4;
    float4*       dst = (float4*)(out + ((size_t)b * cN + n) * cD) + d4;
    *dst = *src;
}

// ---------------- launcher ----------------
extern "C" void kernel_launch(void* const* d_in, const int* in_sizes, int n_in,
                              void* d_out, int out_size, void* d_ws, size_t ws_size,
                              hipStream_t stream) {
    const float* q  = (const float*)d_in[0];
    const float* k  = (const float*)d_in[1];
    const float* v  = (const float*)d_in[2];
    const int*   cl = (const int*)d_in[3];

    float* out = (float*)d_out;                           // [B,N,D]
    float* a0  = out + (size_t)cB * cN * cD;              // [B,C]

    // workspace layout (16B alignment maintained for bf16 uint4 loads)
    const size_t centerElems = (size_t)cB * cC * cD;      // 4,202,496
    float* vout    = (float*)d_ws;                        // B*C*D f32
    float* countsF = vout + centerElems;                  // B0*C f32
    int*   counts  = (int*)(countsF + (size_t)cB0 * cC);
    int*   offs    = counts + cB0 * cC;
    int*   cursor  = offs + cB0 * cC;
    int*   bucket  = cursor + cB0 * cC;                   // B0*N ints
    unsigned short* qcH = (unsigned short*)(bucket + cB0 * cN);   // B*C*D bf16
    unsigned short* kcH = qcH + centerElems;                      // B*C*D bf16
    unsigned short* vcT = kcH + centerElems;                      // B*D*cCp bf16 (transposed)

    // 1) histogram + buckets (deterministic: int atomics + canonical sort at consumer)
    zero_i32_kernel<<<(cB0 * cC + 255) / 256, 256, 0, stream>>>(counts, cB0 * cC);
    count_kernel<<<(cB0 * cN + 255) / 256, 256, 0, stream>>>(cl, counts);
    scan_kernel<<<1, 32, 0, stream>>>(counts, offs, cursor, countsF);
    scatter_kernel<<<(cB0 * cN + 255) / 256, 256, 0, stream>>>(cl, cursor, bucket);
    padzero_kernel<<<(cB * cD * (cCp - cC) + 255) / 256, 256, 0, stream>>>(vcT);

    // 2) cluster means -> bf16 centers (V transposed + padded)
    center_kernel<<<dim3(cC, cB), 128, 0, stream>>>(q, k, v, counts, offs, bucket, qcH, kcH, vcT);

    // 3) clustered attention with count-weighted softmax (WMMA)
    attn_kernel<<<dim3((cC + 15) / 16, cB), 32, 0, stream>>>(qcH, kcH, vcT, countsF, vout, a0);

    // 4) broadcast back to tokens
    gather_kernel<<<(cB * cN * (cD / 4) + 255) / 256, 256, 0, stream>>>(vout, cl, out);
}
